// ScaledDotProductGAT_77360950936036
// MI455X (gfx1250) — matrix operand order repaired and checked by reading, not compiled, not run on hardware
//
#include <hip/hip_runtime.h>

// ---------------- problem constants (fixed by the reference) ----------------
#define N_NODES   50000
#define N_EDGES   800000
#define IN_FEAT   256
#define HEADS     4
#define D_K       32
#define OUT_TOT   384          // Q | K | V concatenated columns (3 * 128)
#define OFF_K     128
#define OFF_V     256
#define M_TILES   3125         // 50000 / 16 exactly
#define N_TILES   24           // 384 / 16
#define NT_PER_WAVE 3          // 8 waves * 3 = 24 N-tiles per block

typedef __attribute__((ext_vector_type(16))) __bf16       v16bf;
typedef __attribute__((ext_vector_type(8)))  float        v8f;
typedef __attribute__((ext_vector_type(8)))  unsigned int v8u;

__device__ __forceinline__ unsigned short f32_to_bf16(float f) {
  unsigned int u = __float_as_uint(f);
  u += 0x7FFFu + ((u >> 16) & 1u);          // round to nearest even
  return (unsigned short)(u >> 16);
}

// ---------------- init: zero out/ssum, set smax to ordered(-inf) -----------
__global__ void __launch_bounds__(256) gat_init(float* __restrict__ out,
                                                unsigned int* __restrict__ smax,
                                                float* __restrict__ ssum) {
  int i = blockIdx.x * 256 + threadIdx.x;
  if (i < N_NODES * (HEADS * D_K)) out[i] = 0.0f;
  if (i < N_NODES * HEADS) {
    smax[i] = 0x007FFFFFu;   // ~bits(-inf) in order-preserving uint encoding
    ssum[i] = 0.0f;
  }
}

// ---------------- f32 -> bf16 conversion of x ------------------------------
__global__ void __launch_bounds__(256) conv_x(const float* __restrict__ x,
                                              unsigned short* __restrict__ xb) {
  int i = blockIdx.x * 256 + threadIdx.x;   // exactly N_NODES*IN_FEAT threads
  xb[i] = f32_to_bf16(x[i]);
}

// ---- build Wt[col][k] (bf16, col-major 384x256) from Wq|Wk|Wv (256x128) ---
__global__ void __launch_bounds__(256) conv_w(const float* __restrict__ Wq,
                                              const float* __restrict__ Wk,
                                              const float* __restrict__ Wv,
                                              unsigned short* __restrict__ Wt) {
  int col = blockIdx.x;        // 0..383
  int k   = threadIdx.x;       // 0..255
  const float* W = (col < 128) ? Wq : ((col < 256) ? Wk : Wv);
  int cc = col & 127;
  Wt[col * IN_FEAT + k] = f32_to_bf16(W[k * 128 + cc]);
}

// ---------------- fused QKV projection via bf16 WMMA -----------------------
// One block (8 waves) owns one 16-row M-tile. Each wave computes 3 adjacent
// 16x16 N-tiles, so every A-fragment load feeds 3 V_WMMA_F32_16X16X32_BF16.
// K=256 in 8 steps -> 24 WMMAs per wave. grid = 3125, block = 256.
__global__ void __launch_bounds__(256) gat_proj_wmma(
    const unsigned short* __restrict__ xb,   // [N_NODES][256] bf16
    const unsigned short* __restrict__ Wt,   // [384][256]     bf16 (col-major W)
    float* __restrict__ QKV) {               // [N_NODES][384] f32
  const int lane   = threadIdx.x & 31;
  const int wave   = threadIdx.x >> 5;
  const int tileM  = blockIdx.x;                  // 0..3124
  const int tileN0 = wave * NT_PER_WAVE;          // 0,3,6,...,21
  const int half   = lane >> 4;                   // 0: lanes 0-15, 1: 16-31
  const int row    = tileM * 16 + (lane & 15);    // A-matrix M index
  const int nlane  = lane & 15;                   // B/D-matrix N index in tile

  const unsigned int* A32 = (const unsigned int*)xb;  // pairs of bf16
  const unsigned int* B32 = (const unsigned int*)Wt;

  v8f acc[NT_PER_WAVE] = {};
#pragma unroll
  for (int ks = 0; ks < 8; ++ks) {               // K = ks*32 .. ks*32+31
    // ---- A fragment (shared by the 3 N-tiles of this wave) ----
    v8u au;
#pragma unroll
    for (int i = 0; i < 8; ++i) {
      // ISA A layout (16-bit, 16x32): VGPR i holds K = {kk, kk+1}
      int kk = (i >> 2) * 16 + (i & 3) * 2 + half * 8;
      au[i] = A32[row * (IN_FEAT / 2) + ks * 16 + (kk >> 1)];
    }
    v16bf a = __builtin_bit_cast(v16bf, au);

#pragma unroll
    for (int j = 0; j < NT_PER_WAVE; ++j) {
      int col = (tileN0 + j) * 16 + nlane;
      v8u bu;
#pragma unroll
      for (int v = 0; v < 8; ++v) {
        // B layout matches A: VGPR v holds K = {kk, kk+1}, kk = (v/4)*16 + (v%4)*2 + half*8
        int kk = (v >> 2) * 16 + (v & 3) * 2 + half * 8;
        bu[v] = B32[col * (IN_FEAT / 2) + ks * 16 + (kk >> 1)];
      }
      v16bf b = __builtin_bit_cast(v16bf, bu);
      acc[j] = __builtin_amdgcn_wmma_f32_16x16x32_bf16(false, a, false, b,
                                                       (short)0, acc[j],
                                                       false, false);
    }
  }
  // D layout: VGPR r -> M = r + 8*half, N = col
#pragma unroll
  for (int j = 0; j < NT_PER_WAVE; ++j) {
    int col = (tileN0 + j) * 16 + nlane;
#pragma unroll
    for (int r = 0; r < 8; ++r) {
      int mm = r + half * 8;
      QKV[(size_t)(tileM * 16 + mm) * OUT_TOT + col] = acc[j][r];
    }
  }
}

// ---------------- per-edge scaled dot scores + segment max -----------------
__global__ void __launch_bounds__(256) gat_scores(
    const float* __restrict__ QKV, const long long* __restrict__ edge,
    float* __restrict__ scores, unsigned int* __restrict__ smax) {
  int idx = blockIdx.x * 256 + threadIdx.x;       // exactly N_EDGES*HEADS
  int e = idx >> 2, h = idx & 3;
  int r = (int)edge[e];
  int c = (int)edge[N_EDGES + e];
  const float4* q = (const float4*)(QKV + (size_t)r * OUT_TOT + h * D_K);
  const float4* k = (const float4*)(QKV + (size_t)c * OUT_TOT + OFF_K + h * D_K);
  float s = 0.0f;
#pragma unroll
  for (int i = 0; i < 8; ++i) {
    float4 qa = q[i], ka = k[i];
    s += qa.x * ka.x + qa.y * ka.y + qa.z * ka.z + qa.w * ka.w;
  }
  s *= 0.17677669529663689f;                      // 1/sqrt(32)
  scores[idx] = s;
  unsigned int b = __float_as_uint(s);
  unsigned int ord = (b & 0x80000000u) ? ~b : (b | 0x80000000u);
  atomicMax(&smax[r * HEADS + h], ord);
}

// ---------------- exp(s - max) + segment sum -------------------------------
__global__ void __launch_bounds__(256) gat_expsum(
    float* __restrict__ scores, const long long* __restrict__ edge,
    const unsigned int* __restrict__ smax, float* __restrict__ ssum) {
  int idx = blockIdx.x * 256 + threadIdx.x;
  int e = idx >> 2, h = idx & 3;
  int r = (int)edge[e];
  unsigned int u = smax[r * HEADS + h];
  float m = (u & 0x80000000u) ? __uint_as_float(u & 0x7FFFFFFFu)
                              : __uint_as_float(~u);
  float ex = __expf(scores[idx] - m);
  scores[idx] = ex;                               // reuse buffer for exp values
  atomicAdd(&ssum[r * HEADS + h], ex);
}

// ---------------- scatter alpha * V[col] into out --------------------------
__global__ void __launch_bounds__(256) gat_scatter(
    const float* __restrict__ ex, const long long* __restrict__ edge,
    const float* __restrict__ ssum, const float* __restrict__ QKV,
    float* __restrict__ out) {
  int idx = blockIdx.x * 256 + threadIdx.x;
  int e = idx >> 2, h = idx & 3;
  int r = (int)edge[e];
  int c = (int)edge[N_EDGES + e];
  float alpha = ex[idx] / ssum[r * HEADS + h];
  const float* v = QKV + (size_t)c * OUT_TOT + OFF_V + h * D_K;
  float* o = out + (size_t)r * (HEADS * D_K) + h * D_K;
#pragma unroll
  for (int d = 0; d < D_K; ++d) atomicAdd(&o[d], alpha * v[d]);
}

// ---------------------------------------------------------------------------
static inline size_t alignup(size_t v) { return (v + 255) & ~(size_t)255; }

extern "C" void kernel_launch(void* const* d_in, const int* in_sizes, int n_in,
                              void* d_out, int out_size, void* d_ws, size_t ws_size,
                              hipStream_t stream) {
  const float*     x    = (const float*)d_in[0];
  const float*     Wq   = (const float*)d_in[1];
  const float*     Wk   = (const float*)d_in[2];
  const float*     Wv   = (const float*)d_in[3];
  const long long* edge = (const long long*)d_in[4];
  float* out = (float*)d_out;

  // workspace layout
  char* ws = (char*)d_ws;
  size_t o = 0;
  unsigned short* xb  = (unsigned short*)(ws + o); o += alignup((size_t)N_NODES * IN_FEAT * 2);
  unsigned short* Wt  = (unsigned short*)(ws + o); o += alignup((size_t)OUT_TOT * IN_FEAT * 2);
  float*          QKV = (float*)(ws + o);          o += alignup((size_t)N_NODES * OUT_TOT * 4);
  float*          sc  = (float*)(ws + o);          o += alignup((size_t)N_EDGES * HEADS * 4);
  unsigned int*   smx = (unsigned int*)(ws + o);   o += alignup((size_t)N_NODES * HEADS * 4);
  float*          ssm = (float*)(ws + o);          o += alignup((size_t)N_NODES * HEADS * 4);

  // 1) zero accumulators (d_out is NOT re-poisoned between replays)
  gat_init<<<(N_NODES * 128 + 255) / 256, 256, 0, stream>>>(out, smx, ssm);

  // 2) bf16 conversion
  conv_x<<<(N_NODES * IN_FEAT) / 256, 256, 0, stream>>>(x, xb);
  conv_w<<<OUT_TOT, IN_FEAT, 0, stream>>>(Wq, Wk, Wv, Wt);

  // 3) fused QKV projection via WMMA (1 block per M-tile, 3 N-tiles per wave)
  gat_proj_wmma<<<M_TILES, 256, 0, stream>>>(xb, Wt, QKV);

  // 4) edge scores + segment max
  const int egrid = (N_EDGES * HEADS) / 256;       // 12500 exactly
  gat_scores<<<egrid, 256, 0, stream>>>(QKV, edge, sc, smx);

  // 5) exp + segment sum
  gat_expsum<<<egrid, 256, 0, stream>>>(sc, edge, smx, ssm);

  // 6) normalize + scatter-add weighted V
  gat_scatter<<<egrid, 256, 0, stream>>>(sc, edge, ssm, QKV, out);

  (void)in_sizes; (void)n_in; (void)out_size; (void)ws_size;
}
